// PointnetSAModuleMSG_7438883356922
// MI455X (gfx1250) — matrix operand order repaired and checked by reading, not compiled, not run on hardware
//
#include <hip/hip_runtime.h>
#include <hip/hip_bf16.h>

#define B_  8
#define N_  4096
#define CF  64
#define NP  1024

typedef __attribute__((ext_vector_type(16))) __bf16 v16bf;
typedef __attribute__((ext_vector_type(8)))  float  v8f;

__device__ __forceinline__ __bf16 f2bf(float f) {
    union { float f; unsigned u; } v; v.f = f;
    unsigned r = v.u + 0x7FFFu + ((v.u >> 16) & 1u);          // round-to-nearest-even
    union { unsigned short s; __bf16 b; } o; o.s = (unsigned short)(r >> 16);
    return o.b;
}

union Frag { uint4 q[2]; __bf16 h[16]; v16bf v; };

// ---------------------------------------------------------------------------
// 1. Furthest point sampling: one workgroup per batch, xyz staged in LDS,
//    per-thread running min-dist in registers, LDS tree argmax per iteration.
// ---------------------------------------------------------------------------
__global__ __launch_bounds__(256) void fps_kernel(const float* __restrict__ xyz,
                                                  int* __restrict__ fpidx,
                                                  float* __restrict__ newxyz,
                                                  float* __restrict__ out_xyz) {
    __shared__ float sx[N_], sy[N_], sz[N_];
    __shared__ float red_v[256];
    __shared__ int   red_i[256];
    __shared__ float lastp[3];
    const int b = blockIdx.x, t = threadIdx.x;
    const float* base = xyz + (size_t)b * N_ * 3;
    for (int i = t; i < N_; i += 256) {
        sx[i] = base[i * 3 + 0];
        sy[i] = base[i * 3 + 1];
        sz[i] = base[i * 3 + 2];
    }
    __syncthreads();
    float d[16];
#pragma unroll
    for (int i = 0; i < 16; ++i) d[i] = 1e10f;
    if (t == 0) {
        fpidx[b * NP] = 0;
        lastp[0] = sx[0]; lastp[1] = sy[0]; lastp[2] = sz[0];
        float* o  = newxyz + (size_t)b * NP * 3;
        float* oo = out_xyz + (size_t)b * NP * 3;
        o[0] = oo[0] = sx[0]; o[1] = oo[1] = sy[0]; o[2] = oo[2] = sz[0];
    }
    __syncthreads();
    for (int it = 1; it < NP; ++it) {
        const float lx = lastp[0], ly = lastp[1], lz = lastp[2];
        float bv = -1.0f; int bi = 0;
#pragma unroll
        for (int i = 0; i < 16; ++i) {
            const int p = t + i * 256;
            const float dx = sx[p] - lx, dy = sy[p] - ly, dz = sz[p] - lz;
            const float dd = dx * dx + dy * dy + dz * dz;
            d[i] = fminf(d[i], dd);
            if (d[i] > bv) { bv = d[i]; bi = p; }
        }
        red_v[t] = bv; red_i[t] = bi;
        __syncthreads();
        for (int off = 128; off > 0; off >>= 1) {
            if (t < off && red_v[t + off] > red_v[t]) {
                red_v[t] = red_v[t + off]; red_i[t] = red_i[t + off];
            }
            __syncthreads();
        }
        if (t == 0) {
            const int w = red_i[0];
            fpidx[b * NP + it] = w;
            lastp[0] = sx[w]; lastp[1] = sy[w]; lastp[2] = sz[w];
            float* o  = newxyz + ((size_t)b * NP + it) * 3;
            float* oo = out_xyz + ((size_t)b * NP + it) * 3;
            o[0] = oo[0] = sx[w]; o[1] = oo[1] = sy[w]; o[2] = oo[2] = sz[w];
        }
        __syncthreads();
    }
}

// ---------------------------------------------------------------------------
// 2. Ball query: one wave32 per (b,p); ballot + prefix popcount emits the
//    first-nsample in-ball indices in ascending index order, padded with the
//    first hit (pointnet2 semantics).
// ---------------------------------------------------------------------------
__global__ __launch_bounds__(256) void ballquery_kernel(const float* __restrict__ xyz,
                                                        const float* __restrict__ newxyz,
                                                        int* __restrict__ idxout,
                                                        float r2, int nsample) {
    const int wid  = (int)((blockIdx.x * (size_t)blockDim.x + threadIdx.x) >> 5);
    const int lane = threadIdx.x & 31;
    const int b = wid / NP, p = wid % NP;
    if (b >= B_) return;
    const float* base = xyz + (size_t)b * N_ * 3;
    const float* c = newxyz + ((size_t)b * NP + p) * 3;
    const float cx = c[0], cy = c[1], cz = c[2];
    int* out = idxout + ((size_t)b * NP + p) * nsample;
    int fill = 0, first = 0, haveFirst = 0;
    for (int bp = 0; bp < N_ && fill < nsample; bp += 32) {
        const int j = bp + lane;
        const float dx = base[j * 3 + 0] - cx;
        const float dy = base[j * 3 + 1] - cy;
        const float dz = base[j * 3 + 2] - cz;
        const bool in = (dx * dx + dy * dy + dz * dz) < r2;
        const unsigned m = (unsigned)__ballot(in);
        if (!haveFirst && m) { first = bp + __ffs((int)m) - 1; haveFirst = 1; }
        const int pos = fill + __popc(m & ((1u << lane) - 1u));
        if (in && pos < nsample) out[pos] = j;
        fill += __popc(m);
    }
    const int cnt = fill < nsample ? fill : nsample;
    for (int pos = cnt + lane; pos < nsample; pos += 32) out[pos] = first;
}

// ---------------------------------------------------------------------------
// 3. Transpose features (B,C,N) -> (B,N,C) so gathers read contiguous rows.
// ---------------------------------------------------------------------------
__global__ void transpose_feats(const float* __restrict__ f, float* __restrict__ ft) {
    __shared__ float tile[32][33];
    const int b = blockIdx.z;
    const int n0 = blockIdx.x * 32, c0 = blockIdx.y * 32;
    const int tx = threadIdx.x, ty = threadIdx.y;     // 32 x 8
#pragma unroll
    for (int i = 0; i < 32; i += 8)
        tile[ty + i][tx] = f[((size_t)b * CF + (c0 + ty + i)) * N_ + n0 + tx];
    __syncthreads();
#pragma unroll
    for (int i = 0; i < 32; i += 8)
        ft[((size_t)b * N_ + (n0 + ty + i)) * CF + c0 + tx] = tile[tx][ty + i];
}

// ---------------------------------------------------------------------------
// 4. Gather: build X0 rows [relxyz(3) | feats(64) | zero-pad to 96] in bf16.
// ---------------------------------------------------------------------------
__global__ __launch_bounds__(256) void gather_kernel(const float* __restrict__ xyz,
                                                     const float* __restrict__ newxyz,
                                                     const float* __restrict__ ft,
                                                     const int* __restrict__ idx,
                                                     __bf16* __restrict__ X0, int S) {
    const size_t r = (size_t)blockIdx.x * 256 + threadIdx.x;
    const size_t R = (size_t)B_ * NP * S;
    if (r >= R) return;
    const size_t bp = r / S;
    const int p = (int)(bp % NP), b = (int)(bp / NP);
    const int j = idx[r];
    const float* c = newxyz + ((size_t)b * NP + p) * 3;
    const float* q = xyz + ((size_t)b * N_ + j) * 3;
    union { __bf16 h[96]; uint4 qv[12]; } row;
    row.h[0] = f2bf(q[0] - c[0]);
    row.h[1] = f2bf(q[1] - c[1]);
    row.h[2] = f2bf(q[2] - c[2]);
    const float4* fr = (const float4*)(ft + ((size_t)b * N_ + j) * CF);
#pragma unroll
    for (int k = 0; k < 16; ++k) {
        const float4 v = fr[k];
        row.h[3 + 4 * k + 0] = f2bf(v.x);
        row.h[3 + 4 * k + 1] = f2bf(v.y);
        row.h[3 + 4 * k + 2] = f2bf(v.z);
        row.h[3 + 4 * k + 3] = f2bf(v.w);
    }
#pragma unroll
    for (int k = 67; k < 96; ++k) row.h[k] = f2bf(0.0f);
    uint4* dst = (uint4*)(X0 + r * 96);
#pragma unroll
    for (int k = 0; k < 12; ++k) dst[k] = row.qv[k];
}

// ---------------------------------------------------------------------------
// 5. Weight prep: (cout,cin) fp32 -> (cout,kpad) bf16, K zero-padded.
// ---------------------------------------------------------------------------
__global__ void wprep_kernel(const float* __restrict__ w, __bf16* __restrict__ wb,
                             int cout, int cin, int kpad) {
    const int i = blockIdx.x * 256 + threadIdx.x;
    if (i >= cout * kpad) return;
    const int o = i / kpad, k = i % kpad;
    wb[i] = (k < cin) ? f2bf(w[o * cin + k]) : f2bf(0.0f);
}

__global__ void zero_kernel(float* p, int n) {
    const int i = blockIdx.x * 256 + threadIdx.x;
    if (i < n) p[i] = 0.0f;
}

// ---------------------------------------------------------------------------
// 6. WMMA GEMM, register-tiled: 8 waves/block, wave = one 16-row M-tile with
//    ALL COUT/16 accumulators held live. Outer loop = K-step: each A fragment
//    (and the fused GroupNorm-affine + ReLU + bf16 convert when FUSED) is
//    built exactly ONCE per K-step, then swept against every N-tile via
//    v_wmma_f32_16x16x32_bf16 (fp32 accumulate).
// ---------------------------------------------------------------------------
template <int COUT, int KPAD, bool FUSED>
__global__ __launch_bounds__(256) void gemm_kernel(const void* __restrict__ Ain,
                                                   const __bf16* __restrict__ Wb,
                                                   const float* __restrict__ coeff,
                                                   float* __restrict__ Hout, int MperB) {
    constexpr int NT = COUT / 16;
    __shared__ float s_a[KPAD], s_c[KPAD];
    const int t = threadIdx.x;
    const int lane = t & 31, wave = t >> 5;
    const size_t mblock = (size_t)blockIdx.x * 128;
    if (FUSED) {
        const int b = (int)(mblock / MperB);
        for (int k = t; k < KPAD; k += 256) {
            s_a[k] = coeff[(b * 256 + k) * 2 + 0];
            s_c[k] = coeff[(b * 256 + k) * 2 + 1];
        }
        __syncthreads();
    }
    const size_t m0 = mblock + (size_t)wave * 16;
    const bool hi = lane >= 16;
    const size_t mrow = m0 + (lane & 15);
    const float*  Af = (const float*)Ain;
    const __bf16* Ab = (const __bf16*)Ain;

    v8f acc[NT];
    const v8f vzero = {};
#pragma unroll
    for (int nt = 0; nt < NT; ++nt) acc[nt] = vzero;

    for (int kb = 0; kb < KPAD; kb += 32) {
        Frag a;
        const int ko = kb + (hi ? 8 : 0);
        if (FUSED) {
            const float* ar = Af + mrow * KPAD;
            const float4 x0 = *(const float4*)(ar + ko);
            const float4 x1 = *(const float4*)(ar + ko + 4);
            const float4 x2 = *(const float4*)(ar + ko + 16);
            const float4 x3 = *(const float4*)(ar + ko + 20);
            const float xs[16] = {x0.x, x0.y, x0.z, x0.w, x1.x, x1.y, x1.z, x1.w,
                                  x2.x, x2.y, x2.z, x2.w, x3.x, x3.y, x3.z, x3.w};
#pragma unroll
            for (int i = 0; i < 8; ++i) {
                const int k = ko + i;
                a.h[i] = f2bf(fmaxf(s_a[k] * xs[i] + s_c[k], 0.0f));
            }
#pragma unroll
            for (int i = 0; i < 8; ++i) {
                const int k = ko + 16 + i;
                a.h[8 + i] = f2bf(fmaxf(s_a[k] * xs[8 + i] + s_c[k], 0.0f));
            }
        } else {
            a.q[0] = *(const uint4*)(Ab + mrow * KPAD + ko);
            a.q[1] = *(const uint4*)(Ab + mrow * KPAD + ko + 16);
        }
#pragma unroll
        for (int nt = 0; nt < NT; ++nt) {
            Frag bf;
            const int n = nt * 16 + (lane & 15);
            const __bf16* wr = Wb + (size_t)n * KPAD + kb + (hi ? 8 : 0);
            bf.q[0] = *(const uint4*)(wr);
            bf.q[1] = *(const uint4*)(wr + 16);
            acc[nt] = __builtin_amdgcn_wmma_f32_16x16x32_bf16(false, a.v, false, bf.v,
                                                              (short)0, acc[nt], false, false);
        }
    }

    const size_t mstore = m0 + (hi ? 8 : 0);
#pragma unroll
    for (int nt = 0; nt < NT; ++nt) {
        const int n = nt * 16 + (lane & 15);
#pragma unroll
        for (int r = 0; r < 8; ++r)
            Hout[(mstore + r) * COUT + n] = acc[nt][r];
    }
}

// ---------------------------------------------------------------------------
// 7. GroupNorm stats: coalesced per-channel sum/sumsq partials -> global atomics.
// ---------------------------------------------------------------------------
template <int COUT>
__global__ __launch_bounds__(256) void stats_kernel(const float* __restrict__ h,
                                                    float* __restrict__ stats, int MperB) {
    __shared__ float s_sum[COUT], s_sq[COUT];
    const int t = threadIdx.x;
    if (t < COUT) { s_sum[t] = 0.0f; s_sq[t] = 0.0f; }
    __syncthreads();
    const int RP   = 256 / COUT;          // rows processed in parallel
    const int ch   = t % COUT;
    const int rloc = t / COUT;
    const size_t row0 = (size_t)blockIdx.x * 1024;
    const int b = (int)(row0 / MperB);
    float sum = 0.0f, sq = 0.0f;
    for (size_t r = row0 + rloc; r < row0 + 1024; r += RP) {
        const float v = h[r * COUT + ch];
        sum += v; sq += v * v;
    }
    atomicAdd(&s_sum[ch], sum);
    atomicAdd(&s_sq[ch], sq);
    __syncthreads();
    if (t < COUT) {
        atomicAdd(&stats[(b * 256 + t) * 2 + 0], s_sum[t]);
        atomicAdd(&stats[(b * 256 + t) * 2 + 1], s_sq[t]);
    }
}

// mean/var per (b,group) -> per (b,channel) affine: y = a*x + c
__global__ void finalize_kernel(const float* __restrict__ stats,
                                const float* __restrict__ gamma,
                                const float* __restrict__ beta,
                                float* __restrict__ coeff, int COUT, int MperB) {
    const int i = blockIdx.x * 256 + threadIdx.x;
    if (i >= B_ * COUT) return;
    const int b = i / COUT, ch = i % COUT;
    const int cg = COUT / 32;
    const int g0 = (ch / cg) * cg;
    float sum = 0.0f, sq = 0.0f;
    for (int k = 0; k < cg; ++k) {
        sum += stats[(b * 256 + g0 + k) * 2 + 0];
        sq  += stats[(b * 256 + g0 + k) * 2 + 1];
    }
    const float cnt  = (float)MperB * (float)cg;
    const float mu   = sum / cnt;
    const float var  = sq / cnt - mu * mu;
    const float istd = rsqrtf(var + 1e-5f);
    const float a = gamma[ch] * istd;
    coeff[(b * 256 + ch) * 2 + 0] = a;
    coeff[(b * 256 + ch) * 2 + 1] = beta[ch] - mu * a;
}

// ---------------------------------------------------------------------------
// 8. Final: affine + ReLU + max over S, write transposed (B, Ctot, NP).
// ---------------------------------------------------------------------------
template <int COUT>
__global__ __launch_bounds__(256) void pool_kernel(const float* __restrict__ h,
                                                   const float* __restrict__ coeff,
                                                   float* __restrict__ out,
                                                   int S, int chbase) {
    const size_t i = (size_t)blockIdx.x * 256 + threadIdx.x;
    const int c = (int)(i % COUT);
    const size_t bp = i / COUT;
    const int p = (int)(bp % NP), b = (int)(bp / NP);
    if (b >= B_) return;
    const float a  = coeff[(b * 256 + c) * 2 + 0];
    const float cc = coeff[(b * 256 + c) * 2 + 1];
    const float* row = h + ((size_t)b * NP + p) * (size_t)S * COUT + c;
    float m = 0.0f;  // ReLU outputs are >= 0, so max over them is >= 0
    for (int s = 0; s < S; ++s)
        m = fmaxf(m, fmaxf(a * row[(size_t)s * COUT] + cc, 0.0f));
    out[((size_t)(b * 384 + chbase + c)) * NP + p] = m;
}

// ---------------------------------------------------------------------------
// Host orchestration
// ---------------------------------------------------------------------------
template <int C2>
static void run_scale(const float* xyz, const float* newxyz, const float* featsT,
                      int* idx, __bf16* X0, __bf16* wb, float* stats, float* coeff,
                      float* R1, float* R2, float* out_feat,
                      const float* w0, const float* g0, const float* b0,
                      const float* w1, const float* g1, const float* b1,
                      const float* w2, const float* g2, const float* b2,
                      int S, float radius, int chbase, hipStream_t stream) {
    const int Mtot = B_ * NP * S;
    const int MperB = NP * S;
    const float r2 = radius * radius;

    ballquery_kernel<<<(B_ * NP * 32) / 256, 256, 0, stream>>>(xyz, newxyz, idx, r2, S);
    gather_kernel<<<(Mtot + 255) / 256, 256, 0, stream>>>(xyz, newxyz, featsT, idx, X0, S);

    // layer 0: 96(pad of 67) -> 64, raw bf16 input
    wprep_kernel<<<(64 * 96 + 255) / 256, 256, 0, stream>>>(w0, wb, 64, 67, 96);
    gemm_kernel<64, 96, false><<<Mtot / 128, 256, 0, stream>>>(X0, wb, nullptr, R1, MperB);
    zero_kernel<<<(B_ * 512 + 255) / 256, 256, 0, stream>>>(stats, B_ * 512);
    stats_kernel<64><<<Mtot / 1024, 256, 0, stream>>>(R1, stats, MperB);
    finalize_kernel<<<(B_ * 64 + 255) / 256, 256, 0, stream>>>(stats, g0, b0, coeff, 64, MperB);

    // layer 1: 64 -> 128, GN+ReLU fused into A-load
    wprep_kernel<<<(128 * 64 + 255) / 256, 256, 0, stream>>>(w1, wb, 128, 64, 64);
    gemm_kernel<128, 64, true><<<Mtot / 128, 256, 0, stream>>>(R1, wb, coeff, R2, MperB);
    zero_kernel<<<(B_ * 512 + 255) / 256, 256, 0, stream>>>(stats, B_ * 512);
    stats_kernel<128><<<Mtot / 1024, 256, 0, stream>>>(R2, stats, MperB);
    finalize_kernel<<<(B_ * 128 + 255) / 256, 256, 0, stream>>>(stats, g1, b1, coeff, 128, MperB);

    // layer 2: 128 -> C2, GN+ReLU fused into A-load
    wprep_kernel<<<(C2 * 128 + 255) / 256, 256, 0, stream>>>(w2, wb, C2, 128, 128);
    gemm_kernel<C2, 128, true><<<Mtot / 128, 256, 0, stream>>>(R2, wb, coeff, R1, MperB);
    zero_kernel<<<(B_ * 512 + 255) / 256, 256, 0, stream>>>(stats, B_ * 512);
    stats_kernel<C2><<<Mtot / 1024, 256, 0, stream>>>(R1, stats, MperB);
    finalize_kernel<<<(B_ * C2 + 255) / 256, 256, 0, stream>>>(stats, g2, b2, coeff, C2, MperB);

    pool_kernel<C2><<<(B_ * NP * C2) / 256, 256, 0, stream>>>(R1, coeff, out_feat, S, chbase);
}

extern "C" void kernel_launch(void* const* d_in, const int* in_sizes, int n_in,
                              void* d_out, int out_size, void* d_ws, size_t ws_size,
                              hipStream_t stream) {
    const float* xyz   = (const float*)d_in[0];
    const float* feats = (const float*)d_in[1];
    auto Wp = [&](int s, int a) { return (const float*)d_in[2 + s * 9 + a * 3]; };
    auto Gp = [&](int s, int a) { return (const float*)d_in[3 + s * 9 + a * 3]; };
    auto Bp = [&](int s, int a) { return (const float*)d_in[4 + s * 9 + a * 3]; };

    char* ws = (char*)d_ws;
    size_t off = 0;
    auto alloc = [&](size_t bytes) { void* p = ws + off; off = (off + bytes + 255) & ~(size_t)255; return p; };

    int*    fpidx  = (int*)   alloc((size_t)B_ * NP * 4);
    float*  newxyz = (float*) alloc((size_t)B_ * NP * 3 * 4);
    float*  featsT = (float*) alloc((size_t)B_ * N_ * CF * 4);
    int*    idx    = (int*)   alloc((size_t)B_ * NP * 64 * 4);
    float*  stats  = (float*) alloc((size_t)B_ * 256 * 2 * 4);
    float*  coeff  = (float*) alloc((size_t)B_ * 256 * 2 * 4);
    __bf16* wb     = (__bf16*)alloc((size_t)256 * 128 * 2);
    __bf16* X0     = (__bf16*)alloc((size_t)B_ * NP * 64 * 96 * 2);      //  96 MB
    float*  R1     = (float*) alloc((size_t)B_ * NP * 64 * 256 * 4);     // 512 MB (h1 / h3)
    float*  R2     = (float*) alloc((size_t)B_ * NP * 64 * 128 * 4);     // 256 MB (h2)
    (void)ws_size; (void)in_sizes; (void)n_in; (void)out_size;

    float* out_xyz  = (float*)d_out;
    float* out_feat = (float*)d_out + (size_t)B_ * NP * 3;

    fps_kernel<<<B_, 256, 0, stream>>>(xyz, fpidx, newxyz, out_xyz);
    transpose_feats<<<dim3(N_ / 32, CF / 32, B_), dim3(32, 8), 0, stream>>>(feats, featsT);

    run_scale<128>(xyz, newxyz, featsT, idx, X0, wb, stats, coeff, R1, R2, out_feat,
                   Wp(0, 0), Gp(0, 0), Bp(0, 0), Wp(0, 1), Gp(0, 1), Bp(0, 1),
                   Wp(0, 2), Gp(0, 2), Bp(0, 2), 32, 0.1f, 0, stream);
    run_scale<256>(xyz, newxyz, featsT, idx, X0, wb, stats, coeff, R1, R2, out_feat,
                   Wp(1, 0), Gp(1, 0), Bp(1, 0), Wp(1, 1), Gp(1, 1), Bp(1, 1),
                   Wp(1, 2), Gp(1, 2), Bp(1, 2), 64, 0.2f, 128, stream);
}
